// GCNHead_76982993814213
// MI455X (gfx1250) — matrix operand order repaired and checked by reading, not compiled, not run on hardware
//
#include <hip/hip_runtime.h>

typedef __attribute__((ext_vector_type(16))) _Float16 v16h;
typedef __attribute__((ext_vector_type(8)))  float    v8f;

#define NN 50000      // nodes  (= 625 * 80)
#define NE 1600000    // edges
#define KIN 512       // in_dim (mult of KT)
#define HD  128       // hidden (= 8 * 16)
#define NC  2
#define NG  8

#define MT    5            // M tiles per wave
#define BROWS (MT * 16)    // 80 rows per block
#define KT    128          // K chunk staged in LDS

// ---------------------------------------------------------------------------
// WMMA GEMM: Out[M x 128] = op(A)[M x K] @ B[K x 128], f16 in, f32 accum.
// 256 threads = 8 waves; wave w owns N-columns [16w,16w+16) and MT=5 M tiles.
// B is staged per-K-chunk into LDS, transposed to [col][kk] f16 so each
// 16-half B fragment is one contiguous 32-byte ds_load. A rows are read
// directly from global (each row belongs to exactly one block) and prefetched
// at the start of the same chunk, before B staging, via global_prefetch_b8.
// Fragment layouts used here, contiguous k in both A and B (wave32):
//   A 16x32 f16 : lane L row M=L%16; halves = kb..kb+15, kb=(L>=16?16:0)
//   B 32x16 f16 : lane L col N=L%16; halves = kb..kb+15, kb=(L>=16?16:0)
//   C/D 16x16 f32: VGPR r, lane L -> M = r + (L>=16?8:0), N = L%16
// ---------------------------------------------------------------------------
template<bool RELU>
__global__ __launch_bounds__(256)
void gemm_wmma_f16(const float* __restrict__ A, const float* __restrict__ B,
                   float* __restrict__ Out, int K) {
  __shared__ __align__(32) _Float16 ldsB[128 * KT];   // [col][kk], 32 KB

  const int tid   = threadIdx.x;
  const int lane  = tid & 31;
  const int wave  = tid >> 5;                 // 0..7 -> N tile
  const int row0  = blockIdx.x * BROWS;       // block M base
  const int l15   = lane & 15;
  const int abase = (lane & 16) ? 16 : 0;     // A K-group base
  const int bbase = (lane & 16) ? 16 : 0;     // B K-group base
  const int col   = wave * 16 + l15;          // output column

  v8f acc[MT] = {};

  for (int kc = 0; kc < K; kc += KT) {
    // Prefetch the A rows of this K chunk (emits global_prefetch_b8).
    for (int off = tid; off < BROWS * (KT / 32); off += 256) {
      int r  = off % BROWS;
      int kq = off / BROWS;
      __builtin_prefetch(&A[(size_t)(row0 + r) * K + kc + kq * 32], 0, 1);
    }
    // Stage B chunk into LDS, transposed + converted to f16.
    // Global reads coalesced (idx -> adjacent cols); LDS stores one-time.
    for (int idx = tid; idx < KT * 128; idx += 256) {
      int kk = idx >> 7;        // 0..KT-1
      int c  = idx & 127;       // column
      ldsB[c * KT + kk] = (_Float16)B[(size_t)(kc + kk) * HD + c];
    }
    __syncthreads();

    for (int k0 = 0; k0 < KT; k0 += 32) {
      const v16h b = *(const v16h*)&ldsB[col * KT + k0 + bbase];
#pragma unroll
      for (int mt = 0; mt < MT; ++mt) {
        const float* ap =
            A + (size_t)(row0 + mt * 16 + l15) * K + (kc + k0 + abase);
        v16h a;
#pragma unroll
        for (int i = 0; i < 8; ++i) {
          float v0 = ap[i];
          float v1 = ap[8 + i];
          if (RELU) { v0 = fmaxf(v0, 0.0f); v1 = fmaxf(v1, 0.0f); }
          a[i]     = (_Float16)v0;
          a[8 + i] = (_Float16)v1;
        }
        acc[mt] = __builtin_amdgcn_wmma_f32_16x16x32_f16(
            /*neg_a=*/false, a, /*neg_b=*/false, b,
            /*c_mod=*/(short)0, acc[mt], /*reuse_a=*/false, /*reuse_b=*/false);
      }
    }
    __syncthreads();   // protect ldsB before next chunk restage
  }

  const int orow0 = row0 + ((lane & 16) ? 8 : 0);
#pragma unroll
  for (int mt = 0; mt < MT; ++mt)
#pragma unroll
    for (int r = 0; r < 8; ++r)
      Out[(size_t)(orow0 + mt * 16 + r) * HD + col] = acc[mt][r];
}

// --------------------------- support kernels -------------------------------
__global__ void k_zero(float* __restrict__ p, int n) {
  int i = blockIdx.x * blockDim.x + threadIdx.x;
  if (i < n) p[i] = 0.0f;
}

__global__ void k_deg(const int* __restrict__ ei, float* __restrict__ deg) {
  int e = blockIdx.x * blockDim.x + threadIdx.x;
  if (e < NE) atomicAdd(&deg[ei[NE + e]], 1.0f);   // dst row of edge_index
}

__global__ void k_dinv(const float* __restrict__ deg, float* __restrict__ dinv) {
  int i = blockIdx.x * blockDim.x + threadIdx.x;
  if (i < NN) dinv[i] = rsqrtf(deg[i] + 1.0f);     // +1 self loop
}

// agg[n][d] = h[n][d] * dinv[n]^2 + bias[d]  (self-loop term + bias init)
__global__ void k_init_agg(const float* __restrict__ h, const float* __restrict__ dinv,
                           const float* __restrict__ bias, float* __restrict__ agg) {
  int idx = blockIdx.x * blockDim.x + threadIdx.x;       // n*32 + dim4
  if (idx >= NN * 32) return;
  int n = idx >> 5, d = (idx & 31) * 4;
  float s = dinv[n]; s *= s;
  const float4 hv = *(const float4*)&h[(size_t)n * HD + d];
  float4 o;
  o.x = hv.x * s + bias[d + 0];
  o.y = hv.y * s + bias[d + 1];
  o.z = hv.z * s + bias[d + 2];
  o.w = hv.w * s + bias[d + 3];
  *(float4*)&agg[(size_t)n * HD + d] = o;
}

// agg[dst] += h[src] * dinv[src]*dinv[dst]   (edge message scatter-add)
__global__ void k_scatter(const int* __restrict__ ei, const float* __restrict__ h,
                          const float* __restrict__ dinv, float* __restrict__ agg) {
  int idx = blockIdx.x * blockDim.x + threadIdx.x;       // e*32 + dim4
  if (idx >= NE * 32) return;
  int e = idx >> 5, d = (idx & 31) * 4;
  int src = ei[e], dst = ei[NE + e];
  float nrm = dinv[src] * dinv[dst];
  const float4 hv = *(const float4*)&h[(size_t)src * HD + d];
  float* ap = &agg[(size_t)dst * HD + d];
  atomicAdd(ap + 0, hv.x * nrm);
  atomicAdd(ap + 1, hv.y * nrm);
  atomicAdd(ap + 2, hv.z * nrm);
  atomicAdd(ap + 3, hv.w * nrm);
}

// sums[batch[n]] += relu(agg[n]); cnts[batch[n]] += 1
__global__ void k_pool(const float* __restrict__ agg, const int* __restrict__ batch,
                       float* __restrict__ sums, float* __restrict__ cnts) {
  int idx = blockIdx.x * blockDim.x + threadIdx.x;       // n*32 + dim4
  if (idx >= NN * 32) return;
  int n = idx >> 5, d = (idx & 31) * 4;
  int g = batch[n];
  const float4 v = *(const float4*)&agg[(size_t)n * HD + d];
  float* sp = &sums[(size_t)g * HD + d];
  atomicAdd(sp + 0, fmaxf(v.x, 0.0f));
  atomicAdd(sp + 1, fmaxf(v.y, 0.0f));
  atomicAdd(sp + 2, fmaxf(v.z, 0.0f));
  atomicAdd(sp + 3, fmaxf(v.w, 0.0f));
  if (d == 0) atomicAdd(&cnts[g], 1.0f);
}

// out[g][c] = (sums[g]/max(cnt,1)) . Wlin[:,c] + blin[c]   (16 values)
__global__ void k_head(const float* __restrict__ sums, const float* __restrict__ cnts,
                       const float* __restrict__ Wlin, const float* __restrict__ blin,
                       float* __restrict__ out) {
  int t = threadIdx.x;
  if (t >= NG * NC) return;
  int g = t / NC, c = t % NC;
  float inv = 1.0f / fmaxf(cnts[g], 1.0f);
  float s = 0.0f;
  for (int j = 0; j < HD; ++j)
    s += sums[(size_t)g * HD + j] * inv * Wlin[(size_t)j * NC + c];
  out[(size_t)g * NC + c] = s + blin[c];
}

// ---------------------------------------------------------------------------
extern "C" void kernel_launch(void* const* d_in, const int* in_sizes, int n_in,
                              void* d_out, int out_size, void* d_ws, size_t ws_size,
                              hipStream_t stream) {
  const float* x     = (const float*)d_in[0];
  const int*   ei    = (const int*)  d_in[1];
  const int*   batch = (const int*)  d_in[2];
  const float* W1    = (const float*)d_in[3];
  const float* b1    = (const float*)d_in[4];
  const float* W2    = (const float*)d_in[5];
  const float* b2    = (const float*)d_in[6];
  const float* Wlin  = (const float*)d_in[7];
  const float* blin  = (const float*)d_in[8];
  float* out = (float*)d_out;

  float* ws   = (float*)d_ws;
  float* h    = ws;                                 // [NN, HD]
  float* agg  = h    + (size_t)NN * HD;             // [NN, HD]
  float* dinv = agg  + (size_t)NN * HD;             // [NN]
  float* degf = dinv + NN;                          // [NN]
  float* sums = degf + NN;                          // [NG, HD]
  float* cnts = sums + (size_t)NG * HD;             // [NG]

  const int B = 256;
  // zero degf + sums + cnts (contiguous region)
  int zn = NN + NG * HD + NG;
  k_zero<<<(zn + B - 1) / B, B, 0, stream>>>(degf, zn);

  // degrees and symmetric norm
  k_deg <<<(NE + B - 1) / B, B, 0, stream>>>(ei, degf);
  k_dinv<<<(NN + B - 1) / B, B, 0, stream>>>(degf, dinv);

  // ---- layer 1: h = x @ W1 (WMMA), agg = h*dinv^2 + b1, scatter edges ----
  gemm_wmma_f16<false><<<NN / BROWS, B, 0, stream>>>(x, W1, h, KIN);
  k_init_agg<<<((size_t)NN * 32 + B - 1) / B, B, 0, stream>>>(h, dinv, b1, agg);
  k_scatter <<<((size_t)NE * 32 + B - 1) / B, B, 0, stream>>>(ei, h, dinv, agg);

  // ---- layer 2: h = relu(agg) @ W2 (WMMA, relu fused into A load) ----
  gemm_wmma_f16<true><<<NN / BROWS, B, 0, stream>>>(agg, W2, h, HD);
  k_init_agg<<<((size_t)NN * 32 + B - 1) / B, B, 0, stream>>>(h, dinv, b2, agg);
  k_scatter <<<((size_t)NE * 32 + B - 1) / B, B, 0, stream>>>(ei, h, dinv, agg);

  // ---- pool (relu fused) + head ----
  k_pool<<<((size_t)NN * 32 + B - 1) / B, B, 0, stream>>>(agg, batch, sums, cnts);
  k_head<<<1, 32, 0, stream>>>(sums, cnts, Wlin, blin, out);
}
